// LogicalGNN_44160853737692
// MI455X (gfx1250) — matrix operand order repaired and checked by reading, not compiled, not run on hardware
//
#include <hip/hip_runtime.h>

#define HIDX 64
#define D1 128          // 2*HID
#define D2 384          // 6*HID
#define NWAVE 2         // waves per workgroup
#define MT 2            // edge tiles per wave (M-blocking for B reuse)

typedef __attribute__((ext_vector_type(16))) _Float16 v16h;
typedef __attribute__((ext_vector_type(8)))  _Float16 v8h;
typedef __attribute__((ext_vector_type(4)))  _Float16 v4h;
typedef __attribute__((ext_vector_type(8)))  float    v8f;

union FragA { v16h v; v8h h8[2]; _Float16 h[16]; };

__device__ __forceinline__ void cvt8(const float* __restrict__ p, _Float16* o) {
  float4 a = *(const float4*)p;
  float4 b = *(const float4*)(p + 4);
  o[0] = (_Float16)a.x; o[1] = (_Float16)a.y; o[2] = (_Float16)a.z; o[3] = (_Float16)a.w;
  o[4] = (_Float16)b.x; o[5] = (_Float16)b.y; o[6] = (_Float16)b.z; o[7] = (_Float16)b.w;
}

__device__ __forceinline__ v8f wmma_f16(v16h a, v16h b, v8f c) {
  // 8 args: (neg_a, A, neg_b, B, c_mod, C, reuse_a, reuse_b)
  return __builtin_amdgcn_wmma_f32_16x16x32_f16(false, a, false, b, (short)0, c, false, false);
}

// Fused per-edge kernel. Each wave owns MT 16-edge WMMA tiles; B fragments are
// loaded once per (nt,kc) and applied to all MT tiles. A2 is fully staged in
// LDS as f16 so GEMM2 A-fragments are pure 16B ds loads.
__global__ __launch_bounds__(NWAVE * 32) void gnn_edge_kernel(
    const float* __restrict__ x,
    const float* __restrict__ rel_table,
    const float* __restrict__ time_table,
    const float* __restrict__ b_rt,
    const float* __restrict__ b_fc,
    const int*   __restrict__ edges,
    const _Float16* __restrict__ Wt_rt,   // [n][k] 128x128 f16 (pre-transposed)
    const _Float16* __restrict__ Wt_fc,   // [n][k] 128x384 f16 (pre-transposed)
    float* __restrict__ outSum,
    int n_edges)
{
  // A2 stage: cols 0..127 = x_src, 128..255 = rel_t, 256..383 = x_dst  (48 KB)
  __shared__ _Float16 As[NWAVE][MT][16][D2];
  __shared__ int      sidx[NWAVE][MT][2][16];   // [src|dst] node ids per row

  const int w  = threadIdx.x >> 5;
  const int l  = threadIdx.x & 31;
  const int hi = l >> 4;        // half-wave K-range select (contiguous order: k = 16*hi + s)
  const int ln = l & 15;        // A row / B,C,D column within tile
  const int base = (blockIdx.x * NWAVE + w) * (16 * MT);
  if (base >= n_edges) return;

  // ---- per-row edge indices (row = ln; clamped, masked at output) ----
  int relI[MT], tsI[MT];
#pragma unroll
  for (int mt = 0; mt < MT; ++mt) {
    int e = base + 16 * mt + ln;
    if (e >= n_edges) e = n_edges - 1;
    if (l < 16) {
      sidx[w][mt][0][ln] = edges[e * 4 + 0];
      sidx[w][mt][1][ln] = edges[e * 4 + 1];
    }
    relI[mt] = edges[e * 4 + 2];
    tsI[mt]  = edges[e * 4 + 3];
  }

  // ---- stage x_src / x_dst rows into LDS (coalesced: whole row per instr) ----
#pragma unroll
  for (int mt = 0; mt < MT; ++mt) {
    for (int r = 0; r < 16; ++r) {
      const int sR = sidx[w][mt][0][r];
      const int dR = sidx[w][mt][1][r];
      float4 a = *(const float4*)(x + (size_t)sR * D1 + 4 * l);
      float4 b = *(const float4*)(x + (size_t)dR * D1 + 4 * l);
      v4h ha, hb;
      ha[0] = (_Float16)a.x; ha[1] = (_Float16)a.y; ha[2] = (_Float16)a.z; ha[3] = (_Float16)a.w;
      hb[0] = (_Float16)b.x; hb[1] = (_Float16)b.y; hb[2] = (_Float16)b.z; hb[3] = (_Float16)b.w;
      *(v4h*)&As[w][mt][r][4 * l]            = ha;   // x_src -> cols 0..127
      *(v4h*)&As[w][mt][r][2 * D1 + 4 * l]   = hb;   // x_dst -> cols 256..383
    }
  }

  // ---------- GEMM1: rel_t = lrelu([rel_e | time_e] @ W_rt + b_rt) ----------
  FragA a1[MT][4];
#pragma unroll
  for (int mt = 0; mt < MT; ++mt) {
    const float* relrow  = rel_table  + relI[mt] * HIDX;
    const float* timerow = time_table + tsI[mt]  * HIDX;
#pragma unroll
    for (int kc = 0; kc < 4; ++kc) {
      const int k0 = kc * 32;
      const int r0 = k0 + 16 * hi;
      const int r1 = k0 + 16 * hi + 8;
      const float* p0 = (r0 < HIDX) ? (relrow + r0) : (timerow + (r0 - HIDX));
      const float* p1 = (r1 < HIDX) ? (relrow + r1) : (timerow + (r1 - HIDX));
      cvt8(p0, a1[mt][kc].h);
      cvt8(p1, a1[mt][kc].h + 8);
    }
  }

  for (int nt = 0; nt < 8; ++nt) {
    v8f acc[MT];
#pragma unroll
    for (int mt = 0; mt < MT; ++mt) acc[mt] = (v8f){};
#pragma unroll
    for (int kc = 0; kc < 4; ++kc) {
      v16h b = *(const v16h*)(Wt_rt + (nt * 16 + ln) * D1 + kc * 32 + 16 * hi);
#pragma unroll
      for (int mt = 0; mt < MT; ++mt)
        acc[mt] = wmma_f16(a1[mt][kc].v, b, acc[mt]);
    }
    const int   n  = nt * 16 + ln;
    const float bn = b_rt[n];
#pragma unroll
    for (int mt = 0; mt < MT; ++mt)
#pragma unroll
      for (int j = 0; j < 8; ++j) {          // D layout: (row = j + 8*hi, col = ln)
        float v = acc[mt][j] + bn;
        v = (v >= 0.0f) ? v : 0.2f * v;
        As[w][mt][8 * hi + j][D1 + n] = (_Float16)v;   // rel_t -> cols 128..255
      }
  }
  // All LDS writes/reads are same-wave; DS ops are in-order per wave.

  // dst node for each of the 8 D-rows this lane owns (per tile)
  int dstRow[MT][8];
#pragma unroll
  for (int mt = 0; mt < MT; ++mt)
#pragma unroll
    for (int j = 0; j < 8; ++j) {
      const int g = base + 16 * mt + 8 * hi + j;
      dstRow[mt][j] = (g < n_edges) ? edges[g * 4 + 1] : -1;
    }

  // ---------- GEMM2: msg = lrelu([x_src | rel_t | x_dst] @ W_fc + b_fc) ----------
  for (int nt = 0; nt < 8; ++nt) {
    v8f acc[MT];
#pragma unroll
    for (int mt = 0; mt < MT; ++mt) acc[mt] = (v8f){};
#pragma unroll
    for (int kc = 0; kc < 12; ++kc) {
      const int k0 = kc * 32;
      v16h b = *(const v16h*)(Wt_fc + (nt * 16 + ln) * D2 + k0 + 16 * hi);
#pragma unroll
      for (int mt = 0; mt < MT; ++mt) {
        FragA ua;
        ua.h8[0] = *(const v8h*)&As[w][mt][ln][k0 + 16 * hi];        // 16B ds load
        ua.h8[1] = *(const v8h*)&As[w][mt][ln][k0 + 16 * hi + 8];   // 16B ds load
        acc[mt] = wmma_f16(ua.v, b, acc[mt]);
      }
    }
    const int   n  = nt * 16 + ln;
    const float bn = b_fc[n];
#pragma unroll
    for (int mt = 0; mt < MT; ++mt)
#pragma unroll
      for (int j = 0; j < 8; ++j) {
        if (dstRow[mt][j] >= 0) {
          float v = acc[mt][j] + bn;
          v = (v >= 0.0f) ? v : 0.2f * v;
          atomicAdd(&outSum[(size_t)dstRow[mt][j] * D1 + n], v);
        }
      }
  }
}

// Pre-transpose + f32->f16 both weight matrices into workspace.
__global__ void gnn_prep(const float* __restrict__ W_rt, const float* __restrict__ W_fc,
                         _Float16* __restrict__ Wt_rt, _Float16* __restrict__ Wt_fc) {
  const int t = blockIdx.x * blockDim.x + threadIdx.x;
  if (t < D1 * D1) {
    const int n = t / D1, k = t % D1;
    Wt_rt[n * D1 + k] = (_Float16)W_rt[k * D1 + n];
  }
  const int t2 = t - D1 * D1;
  if (t2 >= 0 && t2 < D1 * D2) {
    const int n = t2 / D2, k = t2 % D2;
    Wt_fc[n * D2 + k] = (_Float16)W_fc[k * D1 + n];
  }
}

__global__ void gnn_zero(float* __restrict__ out, float* __restrict__ counts, int n_nodes) {
  const int t = blockIdx.x * blockDim.x + threadIdx.x;
  if (t < n_nodes * D1) out[t] = 0.0f;
  if (t < n_nodes)      counts[t] = 0.0f;
}

__global__ void gnn_count(const int* __restrict__ edges, float* __restrict__ counts, int n_edges) {
  const int t = blockIdx.x * blockDim.x + threadIdx.x;
  if (t < n_edges) atomicAdd(&counts[edges[t * 4 + 1]], 1.0f);
}

__global__ void gnn_fin(float* __restrict__ out, const float* __restrict__ counts, int n_nodes) {
  const int t = blockIdx.x * blockDim.x + threadIdx.x;
  if (t < n_nodes * D1) out[t] = out[t] / fmaxf(counts[t >> 7], 1.0f);
}

extern "C" void kernel_launch(void* const* d_in, const int* in_sizes, int n_in,
                              void* d_out, int out_size, void* d_ws, size_t ws_size,
                              hipStream_t stream) {
  const float* x          = (const float*)d_in[0];
  const float* rel_table  = (const float*)d_in[1];
  const float* time_table = (const float*)d_in[2];
  const float* W_rt       = (const float*)d_in[3];
  const float* b_rt       = (const float*)d_in[4];
  const float* W_fc       = (const float*)d_in[5];
  const float* b_fc       = (const float*)d_in[6];
  const int*   edges      = (const int*)d_in[7];

  const int n_nodes = in_sizes[0] / (2 * HIDX);
  const int n_edges = in_sizes[7] / 4;

  // workspace layout: Wt_rt(32KB) | Wt_fc(96KB) | counts(n_nodes*4)
  char* ws = (char*)d_ws;
  _Float16* Wt_rt  = (_Float16*)ws;
  _Float16* Wt_fc  = (_Float16*)(ws + (size_t)D1 * D1 * 2);
  float*    counts = (float*)(ws + (size_t)D1 * D1 * 2 + (size_t)D1 * D2 * 2);
  float*    out    = (float*)d_out;

  const int prepN = D1 * D1 + D1 * D2;
  gnn_prep<<<(prepN + 255) / 256, 256, 0, stream>>>(W_rt, W_fc, Wt_rt, Wt_fc);

  const int zeroN = n_nodes * D1;
  gnn_zero<<<(zeroN + 255) / 256, 256, 0, stream>>>(out, counts, n_nodes);

  gnn_count<<<(n_edges + 255) / 256, 256, 0, stream>>>(edges, counts, n_edges);

  const int edgesPerWG = NWAVE * MT * 16;
  const int wgs = (n_edges + edgesPerWG - 1) / edgesPerWG;
  gnn_edge_kernel<<<wgs, NWAVE * 32, 0, stream>>>(x, rel_table, time_table, b_rt, b_fc,
                                                  edges, Wt_rt, Wt_fc, out, n_edges);

  gnn_fin<<<(n_nodes * D1 + 255) / 256, 256, 0, stream>>>(out, counts, n_nodes);
}
